// RMKVModel_74569222193165
// MI455X (gfx1250) — matrix operand order repaired and checked by reading, not compiled, not run on hardware
//
#include <hip/hip_runtime.h>
#include <hip/hip_bf16.h>

typedef _Float16 h16;
typedef __attribute__((ext_vector_type(8)))  _Float16 v8h;
typedef __attribute__((ext_vector_type(16))) _Float16 v16h;
typedef __attribute__((ext_vector_type(8)))  float    v8f;

static __device__ __forceinline__ v16h mk16(v8h lo, v8h hi) {
  union { v16h v; v8h h[2]; } u;
  u.h[0] = lo; u.h[1] = hi;
  return u.v;
}

constexpr int BB = 2, SS = 2048, MM = 64, DD = 1024;
constexpr int TT = MM + SS;           // 2112
constexpr int HH = 16, HD = 64;
constexpr float NEGM = -1000000000.0f;

// ---------------------------------------------------------------------------
// Elementwise helpers
// ---------------------------------------------------------------------------
__global__ void cast_f16_kernel(const float* __restrict__ src, h16* __restrict__ dst, long n) {
  long i = (long)blockIdx.x * blockDim.x + threadIdx.x;
  if (i < n) dst[i] = (h16)src[i];
}

// x = concat(memory, tokens) along seq, cast to f16: (B, T, D)
__global__ void build_xh_kernel(const float* __restrict__ mem, const float* __restrict__ tok,
                                h16* __restrict__ xh) {
  long i = (long)blockIdx.x * blockDim.x + threadIdx.x;
  if (i >= (long)BB * TT * DD) return;
  int d = i % DD;
  int t = (i / DD) % TT;
  int b = i / ((long)DD * TT);
  float v = (t < MM) ? mem[((long)b * MM + t) * DD + d]
                     : tok[((long)b * SS + (t - MM)) * DD + d];
  xh[i] = (h16)v;
}

// cm[b][t] = 0 for memory rows, attention_mask[b][t-M] for token rows  (B, T)
__global__ void build_cm_kernel(const float* __restrict__ amask, float* __restrict__ cm) {
  long i = (long)blockIdx.x * blockDim.x + threadIdx.x;
  if (i >= (long)BB * TT) return;
  int t = i % TT;
  int b = i / TT;
  cm[i] = (t < MM) ? 0.0f : amask[(size_t)b * SS + (t - MM)];
}

// ---------------------------------------------------------------------------
// Generic WMMA GEMM:  C[M,N] = A[M,K] @ W[N,K]^T  (+bias, +residual)
// block = 128 threads = 4 waves; block tile (64*MT) rows x 64 cols, K step 64.
// W tile (64 x 64 f16 = 8 KB) staged in LDS via async global->LDS copies,
// double-buffered: tile k+1 streams in while 8*MT WMMAs per wave consume tile k.
// grid.x = Mtot/(64*MT), grid.y = N/64.  K must be a multiple of 64.
// ---------------------------------------------------------------------------
template <int MT>
__global__ __launch_bounds__(128)
void wmma_gemm_kernel(const h16* __restrict__ A, const h16* __restrict__ W,
                      const float* __restrict__ bias, const float* __restrict__ R,
                      float* __restrict__ Cf, h16* __restrict__ Ch,
                      int N, int K)
{
  __shared__ __align__(32) h16 wtile[2][64 * 64];

  const int tid  = threadIdx.x;
  const int lane = tid & 31;
  const int wave = tid >> 5;
  const int hf = lane >> 4;     // lane half (0/1)
  const int ql = lane & 15;
  const int row0 = blockIdx.x * (64 * MT) + wave * (16 * MT);
  const int col0 = blockIdx.y * 64;

  // Cooperative async copy of one 64(N) x 64(K) tile of W into LDS buffer `buf`.
  // 512 16-byte chunks; 128 threads handle 4 chunks each.
  auto load_wtile = [&](int k0, int buf) {
#pragma unroll
    for (int cc = 0; cc < 4; ++cc) {
      const int c = tid + cc * 128;
      const int n = c >> 3;          // tile row (output column), 8 chunks per row
      const int j = c & 7;           // 16B chunk within row
      const h16* g = W + (size_t)(col0 + n) * K + k0 + j * 8;
      const unsigned l = (unsigned)(size_t)&wtile[buf][n * 64 + j * 8];
      asm volatile("global_load_async_to_lds_b128 %0, %1, off"
                   :: "v"(l), "v"(g) : "memory");
    }
  };

  load_wtile(0, 0);
  asm volatile("s_wait_asynccnt 0" ::: "memory");
  __syncthreads();

  v8f cf[MT][4] = {};
  const h16* arow[MT];
#pragma unroll
  for (int mt = 0; mt < MT; ++mt)
    arow[mt] = A + (size_t)(row0 + mt * 16 + ql) * K;

  int cur = 0;
  for (int k0 = 0; k0 < K; k0 += 64) {
    if (k0 + 64 < K) load_wtile(k0 + 64, cur ^ 1);
    __builtin_prefetch(arow[0] + k0 + 512, 0, 1);

#pragma unroll
    for (int s = 0; s < 2; ++s) {           // two 32-wide K sub-steps
      const int ks = s * 32;
      v16h b[4];
#pragma unroll
      for (int nf = 0; nf < 4; ++nf) {
        const int n = nf * 16 + ql;
        b[nf] = *(const v16h*)&wtile[cur][n * 64 + ks + 16 * hf];
      }
#pragma unroll
      for (int mt = 0; mt < MT; ++mt) {
        v16h a = *(const v16h*)(arow[mt] + k0 + ks + 16 * hf);
#pragma unroll
        for (int nf = 0; nf < 4; ++nf)
          cf[mt][nf] = __builtin_amdgcn_wmma_f32_16x16x32_f16(
              false, a, false, b[nf], (short)0, cf[mt][nf], false, false);
      }
    }
    asm volatile("s_wait_asynccnt 0" ::: "memory");
    __syncthreads();
    cur ^= 1;
  }

#pragma unroll
  for (int mt = 0; mt < MT; ++mt)
#pragma unroll
    for (int nf = 0; nf < 4; ++nf) {
      const int c = col0 + nf * 16 + ql;
#pragma unroll
      for (int i = 0; i < 8; ++i) {
        const int r = row0 + mt * 16 + i + 8 * hf;
        float v = cf[mt][nf][i];
        if (bias) v += bias[c];
        if (R)    v += R[(size_t)r * N + c];
        if (Cf)   Cf[(size_t)r * N + c] = v;
        if (Ch)   Ch[(size_t)r * N + c] = (h16)v;
      }
    }
}

// ---------------------------------------------------------------------------
// RoPE + repack: qh/kh/vh are (B,T,D) with D = H*hd.
// Outputs: qa/ka (B,H,T,hd) row-major, vt (B,H,hd,T)  (V transposed).
// RoPE applied only to token rows (t >= M), pos = t - M.
// ---------------------------------------------------------------------------
__global__ void rope_pack_kernel(const h16* __restrict__ qh, const h16* __restrict__ kh,
                                 const h16* __restrict__ vh,
                                 h16* __restrict__ qa, h16* __restrict__ ka,
                                 h16* __restrict__ vt)
{
  long idx = (long)blockIdx.x * blockDim.x + threadIdx.x;
  if (idx >= (long)BB * HH * TT * 32) return;
  int i = idx & 31;
  int t = (idx >> 5) % TT;
  long rest = idx / ((long)32 * TT);
  int h = rest % HH;
  int b = rest / HH;

  const size_t src = ((size_t)b * TT + t) * DD + h * HD;
  float q1 = qh[src + i], q2 = qh[src + i + 32];
  float k1 = kh[src + i], k2 = kh[src + i + 32];
  float v1 = vh[src + i], v2 = vh[src + i + 32];

  float rq1 = q1, rq2 = q2, rk1 = k1, rk2 = k2;
  if (t >= MM) {
    float pos = (float)(t - MM);
    float freq = __powf(10000.0f, -(float)i / 32.0f);
    float ang = pos * freq;
    float c = __cosf(ang), s = __sinf(ang);
    rq1 = q1 * c - q2 * s;  rq2 = q1 * s + q2 * c;
    rk1 = k1 * c - k2 * s;  rk2 = k1 * s + k2 * c;
  }
  const size_t bh = (size_t)b * HH + h;
  const size_t dst = (bh * TT + t) * HD;
  qa[dst + i] = (h16)rq1;  qa[dst + i + 32] = (h16)rq2;
  ka[dst + i] = (h16)rk1;  ka[dst + i + 32] = (h16)rk2;
  vt[(bh * HD + i) * TT + t]        = (h16)v1;
  vt[(bh * HD + i + 32) * TT + t]   = (h16)v2;
}

// ---------------------------------------------------------------------------
// Flash attention: one wave (32 threads) per (b, h, 16 token-query rows).
// Streams 32-key tiles: memory keys (2 tiles) + causal token key tiles.
// Online softmax; P staged through LDS to form the WMMA A-fragment.
// ctx written compact as (B, S, D) in f16 (only token queries matter).
// ---------------------------------------------------------------------------
__global__ __launch_bounds__(32)
void flash_attn_kernel(const h16* __restrict__ qa, const h16* __restrict__ ka,
                       const h16* __restrict__ vt, const float* __restrict__ cm,
                       h16* __restrict__ ctxh)
{
  const int lane = threadIdx.x;
  const int hf = lane >> 4;
  const int ql = lane & 15;
  const int blk = blockIdx.x;
  const int qi = blk & 127;          // S/16 = 128 query tiles
  const int h  = (blk >> 7) & 15;
  const int b  = blk >> 11;
  const int p0 = qi * 16;            // first query token position
  const size_t bh = (size_t)b * HH + h;
  const float* cmb = cm + (size_t)b * TT;   // padded per-T mask row

  // Q fragments: rows (M+p0 .. M+p0+15), dims [0,32) and [32,64)
  const h16* qbase = qa + (bh * TT + (MM + p0 + ql)) * HD;
  const v16h aq0 = *(const v16h*)(qbase + 16 * hf);
  const v16h aq1 = *(const v16h*)(qbase + 32 + 16 * hf);

  v8f acc[4] = {};
  float mrun[8], lrun[8], cmq[8];
#pragma unroll
  for (int i = 0; i < 8; ++i) {
    mrun[i] = -1e30f; lrun[i] = 0.0f;
    cmq[i] = cmb[MM + p0 + i + 8 * hf];
  }
  const float scale = 0.125f;        // hd^-0.5
  __shared__ __align__(32) h16 Pt[16 * 32];

  const int nkt = 2 + (qi >> 1) + 1; // 32-key tiles: 2 memory + causal token tiles
  for (int kt = 0; kt < nkt; ++kt) {
    const int k0 = kt * 32;
    v8f sc[2];
#pragma unroll
    for (int cc = 0; cc < 2; ++cc) {
      const h16* kb = ka + (bh * TT + (k0 + cc * 16 + ql)) * HD;
      v16h b0 = *(const v16h*)(kb + 16 * hf);
      v16h b1 = *(const v16h*)(kb + 32 + 16 * hf);
      v8f z = {};
      v8f s = __builtin_amdgcn_wmma_f32_16x16x32_f16(false, aq0, false, b0, (short)0, z, false, false);
      s = __builtin_amdgcn_wmma_f32_16x16x32_f16(false, aq1, false, b1, (short)0, s, false, false);
      sc[cc] = s;
    }

    float pv[2][8], tmax[8];
#pragma unroll
    for (int i = 0; i < 8; ++i) tmax[i] = -1e30f;
#pragma unroll
    for (int cc = 0; cc < 2; ++cc) {
      const int kk = k0 + cc * 16 + ql;            // key T-index for this lane's column
      const float cmk = cmb[kk];                   // unconditional padded mask load
#pragma unroll
      for (int i = 0; i < 8; ++i) {
        const int qpos = p0 + i + 8 * hf;          // query token position
        float mv = cmk + cmq[i];
        if (kk >= MM && (kk - MM) > qpos) mv = NEGM;
        float v = sc[cc][i] * scale + mv;
        pv[cc][i] = v;
        tmax[i] = fmaxf(tmax[i], v);
      }
    }
#pragma unroll
    for (int off = 1; off < 16; off <<= 1)
#pragma unroll
      for (int i = 0; i < 8; ++i)
        tmax[i] = fmaxf(tmax[i], __shfl_xor(tmax[i], off, 32));

    float psum[8];
#pragma unroll
    for (int i = 0; i < 8; ++i) {
      float mnew = fmaxf(mrun[i], tmax[i]);
      float corr = __expf(mrun[i] - mnew);
      lrun[i] *= corr;
      mrun[i] = mnew;
#pragma unroll
      for (int f = 0; f < 4; ++f) acc[f][i] *= corr;
      psum[i] = 0.0f;
    }
#pragma unroll
    for (int cc = 0; cc < 2; ++cc)
#pragma unroll
      for (int i = 0; i < 8; ++i) {
        float e = __expf(pv[cc][i] - mrun[i]);
        psum[i] += e;
        Pt[(i + 8 * hf) * 32 + cc * 16 + ql] = (h16)e;  // 16x32 P tile in LDS
      }
#pragma unroll
    for (int off = 1; off < 16; off <<= 1)
#pragma unroll
      for (int i = 0; i < 8; ++i)
        psum[i] += __shfl_xor(psum[i], off, 32);
#pragma unroll
    for (int i = 0; i < 8; ++i) lrun[i] += psum[i];

    __syncthreads();
    // P (16x32) as WMMA A-fragment
    v16h ap = *(const v16h*)&Pt[ql * 32 + 16 * hf];
    // V fragments from transposed V: lane column = dim, contiguous keys
#pragma unroll
    for (int f = 0; f < 4; ++f) {
      const h16* vb = vt + (bh * HD + (f * 16 + ql)) * TT + k0 + 16 * hf;
      v16h bv = *(const v16h*)vb;
      acc[f] = __builtin_amdgcn_wmma_f32_16x16x32_f16(false, ap, false, bv, (short)0, acc[f], false, false);
    }
    __syncthreads();
  }

  float linv[8];
#pragma unroll
  for (int i = 0; i < 8; ++i) linv[i] = __builtin_amdgcn_rcpf(lrun[i]);
#pragma unroll
  for (int f = 0; f < 4; ++f) {
    const int d = h * HD + f * 16 + ql;
#pragma unroll
    for (int i = 0; i < 8; ++i) {
      const int t = p0 + i + 8 * hf;
      ctxh[((size_t)b * SS + t) * DD + d] = (h16)(acc[f][i] * linv[i]);
    }
  }
}

// ---------------------------------------------------------------------------
// out_tok (B,S,D) f32 -> d_out token rows, + tokens^T f16 (B,D,S) for summary GEMM
// ---------------------------------------------------------------------------
__global__ void finalize_tokens_kernel(const float* __restrict__ outtok,
                                       float* __restrict__ dout,
                                       h16* __restrict__ tokhT)
{
  long i = (long)blockIdx.x * blockDim.x + threadIdx.x;
  if (i >= (long)BB * SS * DD) return;
  int d = i % DD;
  int t = (i / DD) % SS;
  int b = i / ((long)DD * SS);
  float v = outtok[i];
  dout[((size_t)b * TT + MM + t) * DD + d] = v;
  tokhT[((size_t)b * DD + d) * SS + t] = (h16)v;
}

// Softmax over pooling scores: one block per (b, m); scores (B,M,S) f32 -> P (f16)
__global__ __launch_bounds__(256)
void pool_softmax_kernel(const float* __restrict__ scores, const float* __restrict__ amask,
                         h16* __restrict__ Ph)
{
  __shared__ float red[256];
  const int b = blockIdx.x >> 6, m = blockIdx.x & 63;
  const float* srow = scores + ((size_t)b * MM + m) * SS;
  const float* mrow = amask + (size_t)b * SS;
  const float scale = 0.03125f;   // D^-0.5 = 1024^-0.5
  const int tid = threadIdx.x;

  float mx = -1e30f;
  for (int j = tid; j < SS; j += 256) mx = fmaxf(mx, srow[j] * scale + mrow[j]);
  red[tid] = mx; __syncthreads();
  for (int s = 128; s > 0; s >>= 1) {
    if (tid < s) red[tid] = fmaxf(red[tid], red[tid + s]);
    __syncthreads();
  }
  mx = red[0]; __syncthreads();

  float sum = 0.0f;
  for (int j = tid; j < SS; j += 256) sum += __expf(srow[j] * scale + mrow[j] - mx);
  red[tid] = sum; __syncthreads();
  for (int s = 128; s > 0; s >>= 1) {
    if (tid < s) red[tid] += red[tid + s];
    __syncthreads();
  }
  const float inv = __builtin_amdgcn_rcpf(red[0]);
  h16* prow = Ph + ((size_t)b * MM + m) * SS;
  for (int j = tid; j < SS; j += 256)
    prow[j] = (h16)(__expf(srow[j] * scale + mrow[j] - mx) * inv);
}

// comb = concat(memory, summary) f16: (B*M, 2D)
__global__ void build_comb_kernel(const float* __restrict__ mem, const float* __restrict__ summ,
                                  h16* __restrict__ comb)
{
  long i = (long)blockIdx.x * blockDim.x + threadIdx.x;
  if (i >= (long)BB * MM * 2 * DD) return;
  int col = i % (2 * DD);
  long row = i / (2 * DD);
  float v = (col < DD) ? mem[row * DD + col] : summ[row * DD + (col - DD)];
  comb[i] = (h16)v;
}

// Gated memory update -> d_out memory rows
__global__ void gate_kernel(const float* __restrict__ upd, const float* __restrict__ mem,
                            float* __restrict__ dout)
{
  long i = (long)blockIdx.x * blockDim.x + threadIdx.x;
  if (i >= (long)BB * MM * DD) return;
  int d = i % DD;
  int m = (i / DD) % MM;
  int b = i / ((long)DD * MM);
  const size_t row = (size_t)b * MM + m;
  float g = upd[row * 2 * DD + d];
  float c = upd[row * 2 * DD + DD + d];
  float gs = 1.0f / (1.0f + __expf(-g));
  float cs = 1.0f / (1.0f + __expf(-c));
  float nm = mem[i] * gs + (1.0f - gs) * (c * cs);   // siglu(c) = c*sigmoid(c)
  dout[((size_t)b * TT + m) * DD + d] = nm;
}

// ---------------------------------------------------------------------------
extern "C" void kernel_launch(void* const* d_in, const int* in_sizes, int n_in,
                              void* d_out, int out_size, void* d_ws, size_t ws_size,
                              hipStream_t stream)
{
  (void)in_sizes; (void)n_in; (void)out_size; (void)ws_size;
  const float* tokens = (const float*)d_in[0];
  const float* memory = (const float*)d_in[1];
  const float* amask  = (const float*)d_in[2];
  const float* wq = (const float*)d_in[3];
  const float* bq = (const float*)d_in[4];
  const float* wk = (const float*)d_in[5];
  const float* bk = (const float*)d_in[6];
  const float* wv = (const float*)d_in[7];
  const float* bv = (const float*)d_in[8];
  const float* wo = (const float*)d_in[9];
  const float* bo = (const float*)d_in[10];
  const float* poolq = (const float*)d_in[11];
  const float* cellw = (const float*)d_in[12];
  const float* cellb = (const float*)d_in[13];
  float* dout = (float*)d_out;

  char* ws = (char*)d_ws;
  size_t off = 0;
  auto alloc = [&](size_t bytes) -> void* {
    void* p = ws + off;
    off = (off + bytes + 255) & ~(size_t)255;
    return p;
  };

  h16*  xh    = (h16*)alloc((size_t)BB * TT * DD * 2);
  h16*  wqh   = (h16*)alloc((size_t)DD * DD * 2);
  h16*  wkh   = (h16*)alloc((size_t)DD * DD * 2);
  h16*  wvh   = (h16*)alloc((size_t)DD * DD * 2);
  h16*  woh   = (h16*)alloc((size_t)DD * DD * 2);
  h16*  cwh   = (h16*)alloc((size_t)2 * DD * 2 * DD * 2);
  h16*  pqh   = (h16*)alloc((size_t)MM * DD * 2);
  h16*  qh    = (h16*)alloc((size_t)BB * TT * DD * 2);
  h16*  kh    = (h16*)alloc((size_t)BB * TT * DD * 2);
  h16*  vh    = (h16*)alloc((size_t)BB * TT * DD * 2);
  h16*  qa    = (h16*)alloc((size_t)BB * HH * TT * HD * 2);
  h16*  ka    = (h16*)alloc((size_t)BB * HH * TT * HD * 2);
  h16*  vt    = (h16*)alloc((size_t)BB * HH * HD * TT * 2);
  h16*  ctxh  = (h16*)alloc((size_t)BB * SS * DD * 2);
  float* outtok = (float*)alloc((size_t)BB * SS * DD * 4);
  h16*  tokh  = (h16*)alloc((size_t)BB * SS * DD * 2);
  h16*  tokhT = (h16*)alloc((size_t)BB * DD * SS * 2);
  float* pscore = (float*)alloc((size_t)BB * MM * SS * 4);
  h16*  ph    = (h16*)alloc((size_t)BB * MM * SS * 2);
  float* summ = (float*)alloc((size_t)BB * MM * DD * 4);
  h16*  comb  = (h16*)alloc((size_t)BB * MM * 2 * DD * 2);
  float* upd  = (float*)alloc((size_t)BB * MM * 2 * DD * 4);
  float* cmv  = (float*)alloc((size_t)BB * TT * 4);

  auto g1 = [](long n) { return dim3((unsigned)((n + 255) / 256)); };

  // 1) weight casts + x + padded mask
  cast_f16_kernel<<<g1((long)DD * DD), 256, 0, stream>>>(wq, wqh, (long)DD * DD);
  cast_f16_kernel<<<g1((long)DD * DD), 256, 0, stream>>>(wk, wkh, (long)DD * DD);
  cast_f16_kernel<<<g1((long)DD * DD), 256, 0, stream>>>(wv, wvh, (long)DD * DD);
  cast_f16_kernel<<<g1((long)DD * DD), 256, 0, stream>>>(wo, woh, (long)DD * DD);
  cast_f16_kernel<<<g1((long)4 * DD * DD), 256, 0, stream>>>(cellw, cwh, (long)4 * DD * DD);
  cast_f16_kernel<<<g1((long)MM * DD), 256, 0, stream>>>(poolq, pqh, (long)MM * DD);
  build_xh_kernel<<<g1((long)BB * TT * DD), 256, 0, stream>>>(memory, tokens, xh);
  build_cm_kernel<<<g1((long)BB * TT), 256, 0, stream>>>(amask, cmv);

  // 2) QKV projections (Mtot = B*T = 4224, 128-row blocks)
  dim3 gqkv((BB * TT) / 128, DD / 64);
  wmma_gemm_kernel<2><<<gqkv, 128, 0, stream>>>(xh, wqh, bq, nullptr, nullptr, qh, DD, DD);
  wmma_gemm_kernel<2><<<gqkv, 128, 0, stream>>>(xh, wkh, bk, nullptr, nullptr, kh, DD, DD);
  wmma_gemm_kernel<2><<<gqkv, 128, 0, stream>>>(xh, wvh, bv, nullptr, nullptr, vh, DD, DD);

  // 3) RoPE + head-major repack (+ V transpose)
  rope_pack_kernel<<<g1((long)BB * HH * TT * 32), 256, 0, stream>>>(qh, kh, vh, qa, ka, vt);

  // 4) flash attention over token queries only
  flash_attn_kernel<<<BB * HH * (SS / 16), 32, 0, stream>>>(qa, ka, vt, cmv, ctxh);

  // 5) O projection + residual (tokens), Mtot = B*S = 4096
  dim3 go((BB * SS) / 128, DD / 64);
  wmma_gemm_kernel<2><<<go, 128, 0, stream>>>(ctxh, woh, bo, tokens, outtok, tokh, DD, DD);

  // 6) write token rows of d_out, build tokens^T
  finalize_tokens_kernel<<<g1((long)BB * SS * DD), 256, 0, stream>>>(outtok, dout, tokhT);

  // 7) pooling scores: pool_q (64x1024) @ tokens^T -> (64x2048) per batch
  for (int b = 0; b < BB; ++b) {
    dim3 gp(1, SS / 64);
    wmma_gemm_kernel<1><<<gp, 128, 0, stream>>>(pqh, tokh + (size_t)b * SS * DD, nullptr, nullptr,
                                                pscore + (size_t)b * MM * SS, nullptr, SS, DD);
  }

  // 8) softmax over keys
  pool_softmax_kernel<<<BB * MM, 256, 0, stream>>>(pscore, amask, ph);

  // 9) summary: P (64x2048) @ tokens (2048x1024) via tokens^T as W
  for (int b = 0; b < BB; ++b) {
    dim3 gs(1, DD / 64);
    wmma_gemm_kernel<1><<<gs, 128, 0, stream>>>(ph + (size_t)b * MM * SS, tokhT + (size_t)b * DD * SS,
                                                nullptr, nullptr, summ + (size_t)b * MM * DD, nullptr,
                                                DD, SS);
  }

  // 10) cell update
  build_comb_kernel<<<g1((long)BB * MM * 2 * DD), 256, 0, stream>>>(memory, summ, comb);
  dim3 gc((BB * MM) / 128, (2 * DD) / 64);
  wmma_gemm_kernel<2><<<gc, 128, 0, stream>>>(comb, cwh, cellb, nullptr, upd, nullptr, 2 * DD, 2 * DD);

  // 11) gated memory -> d_out memory rows
  gate_kernel<<<g1((long)BB * MM * DD), 256, 0, stream>>>(upd, memory, dout);
}
